// SSMBlock_34840774705405
// MI455X (gfx1250) — matrix operand order repaired and checked by reading, not compiled, not run on hardware
//
#include <hip/hip_runtime.h>
#include <hip/hip_bf16.h>

// ---------------------------------------------------------------------------
// Mamba SSM block for MI455X (gfx1250, wave32, WMMA bf16 + async-to-LDS path)
//   k_gemm_async<EPI> : 128x128 block tile, wave tile 32x64 (2x4 WMMA),
//                       double-buffered global_load_async_to_lds_b128,
//                       branch-free steady-state pipeline.
//                       (GEMM1: xz = xn@W_in, GEMM4: out = g@W_out + res)
//   k_gemm<EPI>       : guarded sync GEMM for ragged shapes
//                       (GEMM2: N=80, GEMM3: K=48)
// ---------------------------------------------------------------------------

#define B_SZ     2
#define L_SZ     1024
#define DMODEL   768
#define DINNER   1536
#define DTRANK   48
#define DSTATE   16
#define XDBL     80              // DTRANK + 2*DSTATE
#define NROWS    (B_SZ * L_SZ)   // 2048
#define N2IN     (2 * DINNER)    // 3072

typedef __attribute__((ext_vector_type(16))) __bf16 bf16x16_t;
typedef __attribute__((ext_vector_type(8)))  float  f32x8_t;

static __device__ inline __bf16 f2bf(float f) {
  unsigned u = __builtin_bit_cast(unsigned, f);
  u += 0x7FFFu + ((u >> 16) & 1u);               // round-to-nearest-even
  unsigned short h = (unsigned short)(u >> 16);
  return __builtin_bit_cast(__bf16, h);
}

// ---------------------------------------------------------------------------
// CDNA5 async global->LDS copy (ASYNCcnt-tracked, bypasses VGPRs).
// Flat LDS pointers carry the LDS byte offset in their low 32 bits.
// ---------------------------------------------------------------------------
static __device__ inline void async_copy_b128(void* lds_dst, const void* gsrc) {
  unsigned l = (unsigned)(uintptr_t)lds_dst;
  asm volatile("global_load_async_to_lds_b128 %0, %1, off"
               :: "v"(l), "v"(gsrc) : "memory");
}
static __device__ inline void wait_async_4() {
  asm volatile("s_wait_asynccnt 0x4" ::: "memory");
}
static __device__ inline void wait_async_0() {
  asm volatile("s_wait_asynccnt 0x0" ::: "memory");
}

// ---------------------------------------------------------------------------
// Weight transpose + convert: src is (K,N) fp32 row-major, dst is (N,K) bf16.
// ---------------------------------------------------------------------------
__global__ void k_transpose_bf16(const float* __restrict__ src,
                                 __bf16* __restrict__ dst, int K, int N) {
  int i = blockIdx.x * 256 + threadIdx.x;
  if (i >= N * K) return;
  int n = i / K, k = i - n * K;
  dst[(long)n * K + k] = f2bf(src[(long)k * N + n]);
}

__global__ void k_negexp(const float* __restrict__ a_log,
                         float* __restrict__ a_neg, int n) {
  int i = blockIdx.x * 256 + threadIdx.x;
  if (i < n) a_neg[i] = -__expf(a_log[i]);
}

// ---------------------------------------------------------------------------
// LayerNorm: one 256-thread block per row, output bf16.
// ---------------------------------------------------------------------------
__global__ __launch_bounds__(256)
void k_layernorm(const float* __restrict__ x, const float* __restrict__ gamma,
                 const float* __restrict__ beta, __bf16* __restrict__ xn,
                 int cols) {
  __shared__ float red[16];
  int row = blockIdx.x;
  const float* xr = x + (long)row * cols;
  float s = 0.f, s2 = 0.f;
  for (int c = threadIdx.x; c < cols; c += 256) {
    float v = xr[c];
    s += v; s2 += v * v;
  }
  #pragma unroll
  for (int off = 16; off > 0; off >>= 1) {
    s  += __shfl_down(s,  off, 32);
    s2 += __shfl_down(s2, off, 32);
  }
  int wid = threadIdx.x >> 5;
  if ((threadIdx.x & 31) == 0) { red[wid] = s; red[8 + wid] = s2; }
  __syncthreads();
  if (threadIdx.x == 0) {
    float a = 0.f, b = 0.f;
    for (int w = 0; w < 8; ++w) { a += red[w]; b += red[8 + w]; }
    red[0] = a; red[8] = b;
  }
  __syncthreads();
  float mean = red[0] / cols;
  float var  = red[8] / cols - mean * mean;
  float rstd = rsqrtf(var + 1e-5f);
  for (int c = threadIdx.x; c < cols; c += 256)
    xn[(long)row * cols + c] = f2bf((xr[c] - mean) * rstd * gamma[c] + beta[c]);
}

// ---------------------------------------------------------------------------
// WMMA tile helpers (16x16x32 bf16, wave32 layouts from ISA 7.12.2)
// ---------------------------------------------------------------------------
#define TM 128
#define TNA 128      // async kernel N tile
#define TN  64       // guarded kernel N tile
#define TK 32

union FragU { bf16x16_t v; uint4 q[2]; };

// A fragment (16x32): lane m=lane&15, kh=lane>>4.
// elems 0..7 -> K = kh*8 + 0..7 ; elems 8..15 -> K = 16 + kh*8 + 0..7
static __device__ inline bf16x16_t ld_fragA(const __bf16* rowbase, int kh) {
  FragU f;
  f.q[0] = *(const uint4*)(rowbase + kh * 8);
  f.q[1] = *(const uint4*)(rowbase + 16 + kh * 8);
  return f.v;
}
// B fragment (32x16, col n per lane): elems 0..7 -> K = kh*8 + 0..7 ; elems 8..15 -> K = 16 + kh*8 + 0..7
static __device__ inline bf16x16_t ld_fragB(const __bf16* rowbase, int kh) {
  FragU f;
  f.q[0] = *(const uint4*)(rowbase + kh * 8);
  f.q[1] = *(const uint4*)(rowbase + 16 + kh * 8);
  return f.v;
}

// Epilogue over a wave's 2 x NTI grid of 16x16 f32 tiles.
// C/D layout: lane n = lane&15, VGPR v -> m = v + 8*(lane>>4)
template <int EPI, int NTI>
static __device__ inline void gemm_epilogue(f32x8_t (&acc)[2][NTI], int lane,
                                            int rowOff, int colOff,
                                            float* C, int N,
                                            const float* bias,
                                            const float* res, __bf16* aux) {
  int half   = lane >> 4;
  int n_lane = lane & 15;
  #pragma unroll
  for (int mt = 0; mt < 2; ++mt)
    #pragma unroll
    for (int nt = 0; nt < NTI; ++nt) {
      f32x8_t c = acc[mt][nt];
      #pragma unroll
      for (int v = 0; v < 8; ++v) {
        int row = rowOff + mt * 16 + v + 8 * half;
        int col = colOff + nt * 16 + n_lane;
        if (col < N) {
          float val = c[v];
          if (EPI == 1) {
            val += bias[col];
            val = (val > 20.f) ? val : log1pf(__expf(val));   // softplus
          }
          if (EPI == 2) val += res[(long)row * N + col];
          C[(long)row * N + col] = val;
          if (EPI == 3 && col < DTRANK)
            aux[(long)row * DTRANK + col] = f2bf(val);
        }
      }
    }
}

// ---------------------------------------------------------------------------
// Unguarded GEMM, async-to-LDS double buffered, branch-free steady state.
// Block tile 128x128, wave tile 32x64 (2x4 WMMA, 6 frag loads per 8 WMMA).
// Requires M%128==0, N%128==0, K%32==0, K/32 >= 2. EPI 0: plain, EPI 2: +res.
// ---------------------------------------------------------------------------
template <int EPI>
__global__ __launch_bounds__(256)
void k_gemm_async(const __bf16* __restrict__ A, int lda,
                  const __bf16* __restrict__ Bt, int ldb,
                  float* __restrict__ C, int M, int N, int K,
                  const float* __restrict__ res) {
  __shared__ __bf16 As[2][TM  * TK];   // 2 x 8 KB
  __shared__ __bf16 Bs[2][TNA * TK];   // 2 x 8 KB

  int tid  = threadIdx.x;
  int lane = tid & 31;
  int wave = tid >> 5;
  int wm = wave >> 1;              // 0..3 : rows of 32
  int wn = wave & 1;               // 0..1 : cols of 64
  int rowBase = blockIdx.y * TM;
  int colBase = blockIdx.x * TNA;

  // copy slots: A = 512 uint4 (2/thread), B = 512 uint4 (2/thread)
  int r0 = tid >> 2, cv = tid & 3;
  int r1 = r0 + 64;
  const __bf16* gA0 = A  + (long)(rowBase + r0) * lda + cv * 8;
  const __bf16* gA1 = A  + (long)(rowBase + r1) * lda + cv * 8;
  const __bf16* gB0 = Bt + (long)(colBase + r0) * ldb + cv * 8;
  const __bf16* gB1 = Bt + (long)(colBase + r1) * ldb + cv * 8;
  int s0 = r0 * TK + cv * 8;
  int s1 = r1 * TK + cv * 8;

  const f32x8_t zero = {0.f, 0.f, 0.f, 0.f, 0.f, 0.f, 0.f, 0.f};
  f32x8_t acc[2][4];
  #pragma unroll
  for (int i = 0; i < 2; ++i)
    #pragma unroll
    for (int j = 0; j < 4; ++j) acc[i][j] = zero;

  int m_lane = lane & 15;
  int kh     = lane >> 4;

  auto issue = [&](int buf, int k0) {
    async_copy_b128(&As[buf][s0], gA0 + k0);
    async_copy_b128(&As[buf][s1], gA1 + k0);
    async_copy_b128(&Bs[buf][s0], gB0 + k0);
    async_copy_b128(&Bs[buf][s1], gB1 + k0);
  };

  auto compute = [&](int buf) {
    bf16x16_t a0 = ld_fragA(&As[buf][(wm * 32 + 0  + m_lane) * TK], kh);
    bf16x16_t a1 = ld_fragA(&As[buf][(wm * 32 + 16 + m_lane) * TK], kh);
    bf16x16_t b[4];
    #pragma unroll
    for (int nt = 0; nt < 4; ++nt)
      b[nt] = ld_fragB(&Bs[buf][(wn * 64 + nt * 16 + m_lane) * TK], kh);
    #pragma unroll
    for (int nt = 0; nt < 4; ++nt) {
      acc[0][nt] = __builtin_amdgcn_wmma_f32_16x16x32_bf16(false, a0, false,
                   b[nt], (short)0, acc[0][nt], false, false);
      acc[1][nt] = __builtin_amdgcn_wmma_f32_16x16x32_bf16(false, a1, false,
                   b[nt], (short)0, acc[1][nt], false, false);
    }
  };

  int KT = K / TK;                 // >= 2 for all uses
  issue(0, 0);
  for (int kt = 0; kt < KT - 1; ++kt) {
    int buf = kt & 1;
    issue(buf ^ 1, (kt + 1) * TK); // prefetch next tile
    wait_async_4();                // previous 4 copies (this tile) landed
    __syncthreads();
    compute(buf);
    __syncthreads();               // readers done before buf reuse at kt+2
  }
  wait_async_0();
  __syncthreads();
  compute((KT - 1) & 1);

  gemm_epilogue<EPI, 4>(acc, lane, rowBase + wm * 32, colBase + wn * 64,
                        C, N, nullptr, res, nullptr);
}

// ---------------------------------------------------------------------------
// Guarded sync GEMM for ragged shapes (zero-fills OOB K/N), 128x64, 2x2.
// EPI 1: softplus(acc+bias[n]), EPI 3: plain + bf16 side-store of cols<48.
// ---------------------------------------------------------------------------
template <int EPI>
__global__ __launch_bounds__(256)
void k_gemm(const __bf16* __restrict__ A, int lda,
            const __bf16* __restrict__ Bt, int ldb,
            float* __restrict__ C, int M, int N, int K,
            const float* __restrict__ bias, __bf16* __restrict__ aux) {
  __shared__ __bf16 As[TM * TK];
  __shared__ __bf16 Bs[TN * TK];

  int tid  = threadIdx.x;
  int lane = tid & 31;
  int wave = tid >> 5;
  int wm = wave >> 1;
  int wn = wave & 1;
  int rowBase = blockIdx.y * TM;
  int colBase = blockIdx.x * TN;

  const f32x8_t zero = {0.f, 0.f, 0.f, 0.f, 0.f, 0.f, 0.f, 0.f};
  f32x8_t acc[2][2];
  acc[0][0] = zero; acc[0][1] = zero; acc[1][0] = zero; acc[1][1] = zero;

  int m_lane = lane & 15;
  int kh     = lane >> 4;

  for (int k0 = 0; k0 < K; k0 += TK) {
    #pragma unroll
    for (int it = 0; it < 2; ++it) {
      int idx = tid + it * 256;
      int r = idx >> 2, cv = idx & 3;
      int kk = k0 + cv * 8;
      uint4 val = make_uint4(0u, 0u, 0u, 0u);
      if (kk < K) val = *(const uint4*)(A + (long)(rowBase + r) * lda + kk);
      *(uint4*)(&As[r * TK + cv * 8]) = val;
    }
    {
      int r = tid >> 2, cv = tid & 3;
      int kk = k0 + cv * 8;
      int nn = colBase + r;
      uint4 val = make_uint4(0u, 0u, 0u, 0u);
      if (kk < K && nn < N) val = *(const uint4*)(Bt + (long)nn * ldb + kk);
      *(uint4*)(&Bs[r * TK + cv * 8]) = val;
    }
    __syncthreads();

    bf16x16_t a0 = ld_fragA(&As[(wm * 32 + 0  + m_lane) * TK], kh);
    bf16x16_t a1 = ld_fragA(&As[(wm * 32 + 16 + m_lane) * TK], kh);
    bf16x16_t b0 = ld_fragB(&Bs[(wn * 32 + 0  + m_lane) * TK], kh);
    bf16x16_t b1 = ld_fragB(&Bs[(wn * 32 + 16 + m_lane) * TK], kh);

    acc[0][0] = __builtin_amdgcn_wmma_f32_16x16x32_bf16(false, a0, false, b0,
                (short)0, acc[0][0], false, false);
    acc[0][1] = __builtin_amdgcn_wmma_f32_16x16x32_bf16(false, a0, false, b1,
                (short)0, acc[0][1], false, false);
    acc[1][0] = __builtin_amdgcn_wmma_f32_16x16x32_bf16(false, a1, false, b0,
                (short)0, acc[1][0], false, false);
    acc[1][1] = __builtin_amdgcn_wmma_f32_16x16x32_bf16(false, a1, false, b1,
                (short)0, acc[1][1], false, false);
    __syncthreads();
  }

  gemm_epilogue<EPI, 2>(acc, lane, rowBase + wm * 32, colBase + wn * 32,
                        C, N, bias, nullptr, aux);
}

// ---------------------------------------------------------------------------
// Causal depthwise conv (K=4) + SiLU. xz is (NROWS, 3072); xc = cols [0,1536).
// ---------------------------------------------------------------------------
__global__ __launch_bounds__(256)
void k_conv_silu(const float* __restrict__ xz, const float* __restrict__ cw,
                 const float* __restrict__ cb, float* __restrict__ xcf,
                 __bf16* __restrict__ xcb) {
  long i = (long)blockIdx.x * 256 + threadIdx.x;
  long total = (long)NROWS * DINNER;
  if (i >= total) return;
  int d = (int)(i % DINNER);
  long bl = i / DINNER;
  int l = (int)(bl % L_SZ);
  int b = (int)(bl / L_SZ);
  const float* w = cw + d * 4;
  float acc = cb[d];
  #pragma unroll
  for (int j = 0; j < 4; ++j) {
    int ll = l - 3 + j;
    if (ll >= 0) acc += w[j] * xz[((long)(b * L_SZ + ll)) * N2IN + d];
  }
  float sv = acc / (1.f + __expf(-acc));           // SiLU
  long r = (long)(b * L_SZ + l);
  xcf[r * DINNER + d] = sv;
  xcb[r * DINNER + d] = f2bf(sv);
}

// ---------------------------------------------------------------------------
// Selective scan: one lane per (b, d); 16 states + 16 A coeffs in registers.
// Fuses D*x skip, SiLU(z) gate, and bf16 store of g = y * silu(z).
// ---------------------------------------------------------------------------
__global__ __launch_bounds__(256)
void k_scan(const float* __restrict__ dt, const float* __restrict__ xdbl,
            const float* __restrict__ xcf, const float* __restrict__ Aneg,
            const float* __restrict__ Dp, const float* __restrict__ xz,
            __bf16* __restrict__ g) {
  int t = blockIdx.x * 256 + threadIdx.x;
  if (t >= B_SZ * DINNER) return;
  int b = t / DINNER;
  int d = t - b * DINNER;

  float Areg[DSTATE], h[DSTATE];
  #pragma unroll
  for (int n = 0; n < DSTATE; ++n) { Areg[n] = Aneg[d * DSTATE + n]; h[n] = 0.f; }
  float Dd = Dp[d];

  for (int l = 0; l < L_SZ; ++l) {
    long r   = (long)b * L_SZ + l;
    float dtv = dt[r * DINNER + d];
    float xv  = xcf[r * DINNER + d];
    const float4* Bp4 = (const float4*)(xdbl + r * XDBL + DTRANK);
    const float4* Cp4 = (const float4*)(xdbl + r * XDBL + DTRANK + DSTATE);
    float y = 0.f;
    #pragma unroll
    for (int q = 0; q < 4; ++q) {
      float4 Bv = Bp4[q];
      float4 Cv = Cp4[q];
      float bx[4] = {Bv.x, Bv.y, Bv.z, Bv.w};
      float cx[4] = {Cv.x, Cv.y, Cv.z, Cv.w};
      #pragma unroll
      for (int j = 0; j < 4; ++j) {
        int n = q * 4 + j;
        float dA = __expf(dtv * Areg[n]);
        h[n] = dA * h[n] + dtv * bx[j] * xv;
        y += h[n] * cx[j];
      }
    }
    y += Dd * xv;
    float zv = xz[r * N2IN + DINNER + d];
    float gz = zv / (1.f + __expf(-zv));           // SiLU gate
    g[r * DINNER + d] = f2bf(y * gz);
  }
}

// ---------------------------------------------------------------------------
// Host launcher
// ---------------------------------------------------------------------------
extern "C" void kernel_launch(void* const* d_in, const int* in_sizes, int n_in,
                              void* d_out, int out_size, void* d_ws,
                              size_t ws_size, hipStream_t stream) {
  const float* x      = (const float*)d_in[0];   // (2,1024,768)
  const float* W_in   = (const float*)d_in[1];   // (768,3072)
  const float* conv_w = (const float*)d_in[2];   // (1536,1,4)
  const float* conv_b = (const float*)d_in[3];   // (1536,)
  const float* W_x    = (const float*)d_in[4];   // (1536,80)
  const float* W_dt   = (const float*)d_in[5];   // (48,1536)
  const float* b_dt   = (const float*)d_in[6];   // (1536,)
  const float* A_log  = (const float*)d_in[7];   // (1536,16)
  const float* D_par  = (const float*)d_in[8];   // (1536,)
  const float* W_out  = (const float*)d_in[9];   // (1536,768)
  const float* gamma  = (const float*)d_in[10];  // (768,)
  const float* beta   = (const float*)d_in[11];  // (768,)
  float* out = (float*)d_out;

  char* p = (char*)d_ws;
  auto take = [&](size_t bytes) {
    char* r = p;
    p += (bytes + 255) & ~size_t(255);
    return r;
  };
  __bf16* Wt_in   = (__bf16*)take((size_t)N2IN   * DMODEL * 2);  // (3072,768)
  __bf16* Wt_x    = (__bf16*)take((size_t)XDBL   * DINNER * 2);  // (80,1536)
  __bf16* Wt_dt   = (__bf16*)take((size_t)DINNER * DTRANK * 2);  // (1536,48)
  __bf16* Wt_out  = (__bf16*)take((size_t)DMODEL * DINNER * 2);  // (768,1536)
  float*  A_neg   = (float*) take((size_t)DINNER * DSTATE * 4);
  __bf16* xn_bf   = (__bf16*)take((size_t)NROWS  * DMODEL * 2);
  float*  xz_f    = (float*) take((size_t)NROWS  * N2IN   * 4);
  float*  xc_f    = (float*) take((size_t)NROWS  * DINNER * 4);
  __bf16* xc_bf   = (__bf16*)take((size_t)NROWS  * DINNER * 2);
  float*  xdbl_f  = (float*) take((size_t)NROWS  * XDBL   * 4);
  __bf16* dtin_bf = (__bf16*)take((size_t)NROWS  * DTRANK * 2);
  float*  dt_f    = (float*) take((size_t)NROWS  * DINNER * 4);
  __bf16* g_bf    = (__bf16*)take((size_t)NROWS  * DINNER * 2);

  auto cdiv = [](long a, long b) { return (int)((a + b - 1) / b); };

  // weights -> transposed bf16, A = -exp(A_log)
  k_transpose_bf16<<<cdiv((long)DMODEL * N2IN, 256),   256, 0, stream>>>(W_in,  Wt_in,  DMODEL, N2IN);
  k_transpose_bf16<<<cdiv((long)DINNER * XDBL, 256),   256, 0, stream>>>(W_x,   Wt_x,   DINNER, XDBL);
  k_transpose_bf16<<<cdiv((long)DTRANK * DINNER, 256), 256, 0, stream>>>(W_dt,  Wt_dt,  DTRANK, DINNER);
  k_transpose_bf16<<<cdiv((long)DINNER * DMODEL, 256), 256, 0, stream>>>(W_out, Wt_out, DINNER, DMODEL);
  k_negexp<<<cdiv((long)DINNER * DSTATE, 256), 256, 0, stream>>>(A_log, A_neg, DINNER * DSTATE);

  // LayerNorm
  k_layernorm<<<NROWS, 256, 0, stream>>>(x, gamma, beta, xn_bf, DMODEL);

  // GEMM1 (async): xz = xn @ W_in
  {
    dim3 grid(N2IN / TNA, NROWS / TM);
    k_gemm_async<0><<<grid, 256, 0, stream>>>(xn_bf, DMODEL, Wt_in, DMODEL,
                                              xz_f, NROWS, N2IN, DMODEL, nullptr);
  }

  // conv + SiLU
  k_conv_silu<<<cdiv((long)NROWS * DINNER, 256), 256, 0, stream>>>(xz_f, conv_w, conv_b, xc_f, xc_bf);

  // GEMM2 (guarded): x_dbl = xc @ W_x (+ bf16 side-store of dt slice)
  {
    dim3 grid(cdiv(XDBL, TN), NROWS / TM);
    k_gemm<3><<<grid, 256, 0, stream>>>(xc_bf, DINNER, Wt_x, DINNER, xdbl_f,
                                        NROWS, XDBL, DINNER, nullptr, dtin_bf);
  }

  // GEMM3 (guarded): dt = softplus(dtin @ W_dt + b_dt)
  {
    dim3 grid(DINNER / TN, NROWS / TM);
    k_gemm<1><<<grid, 256, 0, stream>>>(dtin_bf, DTRANK, Wt_dt, DTRANK, dt_f,
                                        NROWS, DINNER, DTRANK, b_dt, nullptr);
  }

  // selective scan + gate
  k_scan<<<cdiv((long)B_SZ * DINNER, 256), 256, 0, stream>>>(dt_f, xdbl_f, xc_f,
                                                             A_neg, D_par, xz_f, g_bf);

  // GEMM4 (async): out = g @ W_out + residual
  {
    dim3 grid(DMODEL / TNA, NROWS / TM);
    k_gemm_async<2><<<grid, 256, 0, stream>>>(g_bf, DINNER, Wt_out, DINNER,
                                              out, NROWS, DMODEL, DINNER, x);
  }
}
